// MultiHeadedAttention_4784593568456
// MI455X (gfx1250) — matrix operand order repaired and checked by reading, not compiled, not run on hardware
//
#include <hip/hip_runtime.h>

// ---------------- problem constants ----------------
#define Bdim 2
#define Sdim 2048
#define Edim 512      // head dim
#define Hdim 8
#define Fdim 4096     // H*E
#define RSCALE 0.04419417382415922f   // 1/sqrt(512)

// ---------------- WMMA types ----------------
typedef __attribute__((ext_vector_type(16))) __bf16 v16bf;
typedef __attribute__((ext_vector_type(8)))  float  v8f;

union FragBF {
    uint4 u[2];   // 32 bytes
    v16bf v;
};

__device__ __forceinline__ unsigned short f2bf(float f) {
    unsigned int u = __float_as_uint(f);
    u += 0x7FFFu + ((u >> 16) & 1u);          // round to nearest even
    return (unsigned short)(u >> 16);
}

__device__ __forceinline__ v8f wmma_bf16(v16bf a, v16bf b, v8f c) {
    return __builtin_amdgcn_wmma_f32_16x16x32_bf16(
        /*neg_a=*/false, a, /*neg_b=*/false, b,
        /*c_mod=*/(short)0, c, /*reuse_a=*/false, /*reuse_b=*/false);
}

// ---------------- CDNA5 async global->LDS copy ----------------
// ISA 15.18.3: GLOBAL_LOAD_ASYNC_TO_LDS_B128  (tracked with ASYNCcnt)
// VDST = VGPR holding LDS byte address, VADDR = 64-bit global address.
// Low 32 bits of a generic LDS pointer are the wave-relative LDS offset.
__device__ __forceinline__ void async_load_b128(unsigned lds_off, const void* gaddr) {
    asm volatile("global_load_async_to_lds_b128 %0, %1, off"
                 :: "v"(lds_off), "v"(gaddr) : "memory");
}
__device__ __forceinline__ void wait_async0() {
    asm volatile("s_wait_asynccnt 0x0" ::: "memory");
}
__device__ __forceinline__ unsigned lds_off32(const void* p) {
    return (unsigned)(size_t)p;
}

// ---------------- fp32 -> bf16 conversion ----------------
__global__ __launch_bounds__(256) void cvt_bf16_kernel(
    const float* __restrict__ in, unsigned short* __restrict__ out, int n4)
{
    int i = blockIdx.x * blockDim.x + threadIdx.x;
    if (i >= n4) return;
    float4 f = ((const float4*)in)[i];
    ushort4 r;
    r.x = f2bf(f.x); r.y = f2bf(f.y); r.z = f2bf(f.z); r.w = f2bf(f.w);
    ((ushort4*)out)[i] = r;
}

// ---------------- tiled bf16 WMMA GEMM ----------------
// C[M][N] = A[M][K] * B[N][K]^T   (both row-major, K contiguous)
// mode 0: C bf16 row-major      (Q, K projections)
// mode 1: C bf16 written as V^T [B,H,E,S]   (V projection)
// mode 2: C fp32 row-major      (final output projection)
// Double-buffered LDS, filled with async-to-LDS copies; one barrier per K-step.
#define ASTR 40   // padded LDS row stride (elements): 20-bank step -> conflict free
__global__ __launch_bounds__(256) void gemm_bf16_kernel(
    const unsigned short* __restrict__ A, const unsigned short* __restrict__ B,
    void* __restrict__ Cout, int M, int N, int K, int mode)
{
    __shared__ __align__(16) unsigned short As[2][128 * ASTR];
    __shared__ __align__(16) unsigned short Bs[2][128 * ASTR];

    const int tid  = threadIdx.x;
    const int wid  = tid >> 5;
    const int lane = tid & 31;
    const int tM   = blockIdx.y * 128;
    const int tN   = blockIdx.x * 128;
    const int m0   = (wid >> 1) * 32;   // 4 waves along M
    const int n0   = (wid & 1) * 64;    // 2 waves along N

    const v8f vzero = {0.f,0.f,0.f,0.f,0.f,0.f,0.f,0.f};
    v8f acc[2][4];
#pragma unroll
    for (int mf = 0; mf < 2; ++mf)
#pragma unroll
        for (int nf = 0; nf < 4; ++nf) acc[mf][nf] = vzero;

    // 512 uint4 per 128x32 tile; 256 threads issue 2 async copies per tile
    auto issue_tile = [&](int k0, int buf) {
#pragma unroll
        for (int it = 0; it < 2; ++it) {
            int j   = tid + it * 256;
            int row = j >> 2;
            int seg = j & 3;
            async_load_b128(lds_off32(&As[buf][row * ASTR + seg * 8]),
                            A + (size_t)(tM + row) * K + k0 + seg * 8);
            async_load_b128(lds_off32(&Bs[buf][row * ASTR + seg * 8]),
                            B + (size_t)(tN + row) * K + k0 + seg * 8);
        }
    };

    issue_tile(0, 0);
    int buf = 0;
    for (int k0 = 0; k0 < K; k0 += 32) {
        wait_async0();          // this wave's fills for stage `buf` landed in LDS
        __syncthreads();        // everyone's fills visible; prior compute done
        if (k0 + 32 < K) issue_tile(k0 + 32, buf ^ 1);

        FragBF a[2], b[4];
#pragma unroll
        for (int mf = 0; mf < 2; ++mf) {
            int row = m0 + mf * 16 + (lane & 15);
            int kb  = (lane >> 4) * 8;
            a[mf].u[0] = *(const uint4*)&As[buf][row * ASTR + kb];
            a[mf].u[1] = *(const uint4*)&As[buf][row * ASTR + kb + 16];
        }
#pragma unroll
        for (int nf = 0; nf < 4; ++nf) {
            int col = n0 + nf * 16 + (lane & 15);
            int kb  = (lane >> 4) * 8;
            b[nf].u[0] = *(const uint4*)&Bs[buf][col * ASTR + kb];
            b[nf].u[1] = *(const uint4*)&Bs[buf][col * ASTR + kb + 16];
        }
#pragma unroll
        for (int mf = 0; mf < 2; ++mf)
#pragma unroll
            for (int nf = 0; nf < 4; ++nf)
                acc[mf][nf] = wmma_bf16(a[mf].v, b[nf].v, acc[mf][nf]);
        buf ^= 1;
    }

    // epilogue: C frag layout -> VGPR r, lanes<16: M=r, lanes>=16: M=r+8; N=lane&15
    const int hi  = lane >> 4;
    const int nlo = lane & 15;
#pragma unroll
    for (int mf = 0; mf < 2; ++mf)
#pragma unroll
        for (int nf = 0; nf < 4; ++nf)
#pragma unroll
            for (int r = 0; r < 8; ++r) {
                int gm = tM + m0 + mf * 16 + r + 8 * hi;
                int gn = tN + n0 + nf * 16 + nlo;
                float v = acc[mf][nf][r];
                if (mode == 0) {
                    ((unsigned short*)Cout)[(size_t)gm * N + gn] = f2bf(v);
                } else if (mode == 1) {
                    int b_ = gm / Sdim, s_ = gm % Sdim;
                    int h_ = gn / Edim, e_ = gn % Edim;
                    ((unsigned short*)Cout)[((size_t)(b_ * Hdim + h_) * Edim + e_) * Sdim + s_] = f2bf(v);
                } else {
                    ((float*)Cout)[(size_t)gm * N + gn] = v;
                }
            }
}

// ---------------- flash attention ----------------
// Workgroup = 4 waves, one (b, h, 16-query tile). Each wave owns a 128-wide
// slice of the 512-wide head dim. KV tile = 64 keys.
#define QSTR 520   // padded Q LDS stride
#define PSTR 72    // padded P LDS stride
__global__ __launch_bounds__(128) void flash_attn_kernel(
    const unsigned short* __restrict__ Q,   // [B,S,F] bf16 (heads = F slices)
    const unsigned short* __restrict__ Km,  // [B,S,F] bf16
    const unsigned short* __restrict__ VT,  // [B,H,E,S] bf16
    unsigned short* __restrict__ AO)        // [B,S,F] bf16
{
    __shared__ __align__(16) unsigned short Qs[16 * QSTR];
    __shared__ __align__(16) float          sc[4 * 16 * 64];   // per-wave partial scores
    __shared__ __align__(16) unsigned short Pt[16 * PSTR];     // bf16 probabilities
    __shared__ float s_alpha[16];
    __shared__ float s_inv[16];

    const int tid  = threadIdx.x;
    const int wid  = tid >> 5;
    const int lane = tid & 31;
    const int hi   = lane >> 4;
    const int nlo  = lane & 15;

    const int qb = blockIdx.x;   // 16-query tile
    const int h  = blockIdx.y;
    const int bb = blockIdx.z;
    const int q0 = qb * 16;

    const unsigned short* Kbh = Km + (size_t)bb * Sdim * Fdim + h * Edim; // stride Fdim
    const unsigned short* Vbh = VT + (size_t)(bb * Hdim + h) * Edim * Sdim; // [e][s]

    // stage Q tile [16][512] into LDS with async copies
    {
        int row = tid >> 3;
        int cs  = (tid & 7) * 64;
        const unsigned short* src =
            Q + (size_t)(bb * Sdim + q0 + row) * Fdim + h * Edim + cs;
#pragma unroll
        for (int j = 0; j < 8; ++j)
            async_load_b128(lds_off32(&Qs[row * QSTR + cs + j * 8]), src + j * 8);
    }
    wait_async0();
    __syncthreads();

    const v8f vzero = {0.f,0.f,0.f,0.f,0.f,0.f,0.f,0.f};
    v8f o[8];
#pragma unroll
    for (int nf = 0; nf < 8; ++nf) o[nf] = vzero;

    float rmax = -3.0e38f;  // meaningful in threads 0..15
    float rsum = 0.0f;

    const int ktmax = (q0 + 15) >> 6;
    for (int kt = 0; kt <= ktmax; ++kt) {
        // L2 warm-up for the next KV tile (global_prefetch_b8 hints)
        if (kt < ktmax) {
            __builtin_prefetch(Kbh + (size_t)((kt + 1) * 64 + (tid & 63)) * Fdim +
                                   (tid >> 6) * 256, 0, 1);
            __builtin_prefetch(Vbh + (size_t)(tid * 4) * Sdim + (kt + 1) * 64, 0, 1);
        }

        // ---- partial scores over this wave's 128-wide E slice ----
        v8f s[4];
#pragma unroll
        for (int nf = 0; nf < 4; ++nf) s[nf] = vzero;

        const int ebase = wid * 128;
#pragma unroll
        for (int ks = 0; ks < 4; ++ks) {
            int eo = ebase + ks * 32;
            FragBF a;
            {
                int row = nlo;
                int kb  = eo + hi * 16;
                a.u[0] = *(const uint4*)&Qs[row * QSTR + kb];
                a.u[1] = *(const uint4*)&Qs[row * QSTR + kb + 8];
            }
#pragma unroll
            for (int nf = 0; nf < 4; ++nf) {
                int kg = kt * 64 + nf * 16 + nlo;
                const unsigned short* bp = Kbh + (size_t)kg * Fdim + eo + hi * 16;
                FragBF bk;
                bk.u[0] = ((const uint4*)bp)[0];
                bk.u[1] = ((const uint4*)bp)[1];
                s[nf] = wmma_bf16(a.v, bk.v, s[nf]);
            }
        }
        // store partial scores
#pragma unroll
        for (int nf = 0; nf < 4; ++nf)
#pragma unroll
            for (int r = 0; r < 8; ++r)
                sc[(wid * 16 + r + 8 * hi) * 64 + nf * 16 + nlo] = s[nf][r];
        __syncthreads();

        // ---- online softmax: one thread per query row ----
        if (tid < 16) {
            const int m  = tid;
            const int qg = q0 + m;
            float tmax = -3.0e38f;
            for (int n = 0; n < 64; ++n) {
                float v_ = sc[(0 * 16 + m) * 64 + n] + sc[(1 * 16 + m) * 64 + n] +
                           sc[(2 * 16 + m) * 64 + n] + sc[(3 * 16 + m) * 64 + n];
                v_ *= RSCALE;
                if (kt * 64 + n > qg) v_ = -3.0e38f;     // causal mask
                sc[m * 64 + n] = v_;                     // reuse wave-0 region
                tmax = fmaxf(tmax, v_);
            }
            float nm = fmaxf(rmax, tmax);
            float al = __expf(rmax - nm);
            rmax = nm;
            float ps = 0.0f;
            for (int n = 0; n < 64; ++n) {
                float p = __expf(sc[m * 64 + n] - nm);
                ps += p;
                Pt[m * PSTR + n] = f2bf(p);
            }
            rsum = rsum * al + ps;
            s_alpha[m] = al;
        }
        __syncthreads();

        // ---- rescale O, then P*V over this wave's E slice ----
        {
            float av[8];
#pragma unroll
            for (int r = 0; r < 8; ++r) av[r] = s_alpha[r + 8 * hi];
#pragma unroll
            for (int nf = 0; nf < 8; ++nf)
#pragma unroll
                for (int r = 0; r < 8; ++r) o[nf][r] *= av[r];
        }
#pragma unroll
        for (int ks = 0; ks < 2; ++ks) {
            FragBF a;
            {
                int row = nlo;
                int kb  = ks * 32 + hi * 16;
                a.u[0] = *(const uint4*)&Pt[row * PSTR + kb];
                a.u[1] = *(const uint4*)&Pt[row * PSTR + kb + 8];
            }
#pragma unroll
            for (int nf = 0; nf < 8; ++nf) {
                int e = wid * 128 + nf * 16 + nlo;
                const unsigned short* vp =
                    Vbh + (size_t)e * Sdim + kt * 64 + ks * 32 + hi * 16;
                FragBF bv;
                bv.u[0] = ((const uint4*)vp)[0];
                bv.u[1] = ((const uint4*)vp)[1];
                o[nf] = wmma_bf16(a.v, bv.v, o[nf]);
            }
        }
        __syncthreads();   // protect sc / Pt / alpha before next tile
    }

    // ---- epilogue: divide by row sums, write bf16 attn output ----
    if (tid < 16) s_inv[tid] = 1.0f / rsum;
    __syncthreads();
    float iv[8];
#pragma unroll
    for (int r = 0; r < 8; ++r) iv[r] = s_inv[r + 8 * hi];
#pragma unroll
    for (int nf = 0; nf < 8; ++nf)
#pragma unroll
        for (int r = 0; r < 8; ++r) {
            int gm = q0 + r + 8 * hi;
            int gn = h * Edim + wid * 128 + nf * 16 + nlo;
            AO[(size_t)(bb * Sdim + gm) * Fdim + gn] = f2bf(o[nf][r] * iv[r]);
        }
}

// ---------------- host orchestration ----------------
extern "C" void kernel_launch(void* const* d_in, const int* in_sizes, int n_in,
                              void* d_out, int out_size, void* d_ws, size_t ws_size,
                              hipStream_t stream) {
    const float* x  = (const float*)d_in[0];
    // d_in[1] = attn_mask (causal, applied arithmetically in-kernel)
    const float* Wq = (const float*)d_in[2];
    const float* Wk = (const float*)d_in[3];
    const float* Wv = (const float*)d_in[4];
    const float* Wo = (const float*)d_in[5];

    char* ws = (char*)d_ws;
    size_t off = 0;
    unsigned short* xb  = (unsigned short*)(ws + off); off += (size_t)Bdim * Sdim * Edim * 2;
    unsigned short* wqb = (unsigned short*)(ws + off); off += (size_t)Fdim * Edim * 2;
    unsigned short* wkb = (unsigned short*)(ws + off); off += (size_t)Fdim * Edim * 2;
    unsigned short* wvb = (unsigned short*)(ws + off); off += (size_t)Fdim * Edim * 2;
    unsigned short* wob = (unsigned short*)(ws + off); off += (size_t)Edim * Fdim * 2;
    unsigned short* Qb  = (unsigned short*)(ws + off); off += (size_t)Bdim * Sdim * Fdim * 2;
    unsigned short* Kb  = (unsigned short*)(ws + off); off += (size_t)Bdim * Sdim * Fdim * 2;
    unsigned short* VTb = (unsigned short*)(ws + off); off += (size_t)Bdim * Hdim * Edim * Sdim * 2;
    unsigned short* AOb = (unsigned short*)(ws + off); off += (size_t)Bdim * Sdim * Fdim * 2;

    const int M = Bdim * Sdim;   // 4096

    // fp32 -> bf16
    {
        int n4x = (Bdim * Sdim * Edim) / 4;
        int n4w = (Fdim * Edim) / 4;
        cvt_bf16_kernel<<<(n4x + 255) / 256, 256, 0, stream>>>(x,  xb,  n4x);
        cvt_bf16_kernel<<<(n4w + 255) / 256, 256, 0, stream>>>(Wq, wqb, n4w);
        cvt_bf16_kernel<<<(n4w + 255) / 256, 256, 0, stream>>>(Wk, wkb, n4w);
        cvt_bf16_kernel<<<(n4w + 255) / 256, 256, 0, stream>>>(Wv, wvb, n4w);
        cvt_bf16_kernel<<<(n4w + 255) / 256, 256, 0, stream>>>(Wo, wob, n4w);
    }

    // Q/K/V projections
    dim3 gP(Fdim / 128, M / 128);
    gemm_bf16_kernel<<<gP, 256, 0, stream>>>(xb, wqb, Qb,  M, Fdim, Edim, 0);
    gemm_bf16_kernel<<<gP, 256, 0, stream>>>(xb, wkb, Kb,  M, Fdim, Edim, 0);
    gemm_bf16_kernel<<<gP, 256, 0, stream>>>(xb, wvb, VTb, M, Fdim, Edim, 1);

    // flash attention
    dim3 gF(Sdim / 16, Hdim, Bdim);
    flash_attn_kernel<<<gF, 128, 0, stream>>>(Qb, Kb, VTb, AOb);

    // output projection -> fp32 d_out [B,S,E]
    dim3 gO(Edim / 128, M / 128);
    gemm_bf16_kernel<<<gO, 256, 0, stream>>>(AOb, wob, d_out, M, Edim, Fdim, 2);
}
